// GeoDiff_58875411693663
// MI455X (gfx1250) — matrix operand order repaired and checked by reading, not compiled, not run on hardware
//
#include <hip/hip_runtime.h>
#include <hip/hip_bf16.h>
#include <math.h>

// ---------------- problem constants ----------------
#define NB    16      // batches
#define NA    256     // atoms per batch
#define NF    64      // input features
#define NH    256     // hidden
#define NTEMB 32
#define NUMT  100
#define NL    4
#define NE    131072  // edges
#define NN    (NB*NA) // 4096 nodes
#define WPB   2       // waves per block (each wave owns one 16-row M tile)

typedef _Float16 half8  __attribute__((ext_vector_type(8)));
typedef _Float16 v16h   __attribute__((ext_vector_type(16)));
typedef float    v8f    __attribute__((ext_vector_type(8)));

union V16 { v16h v; half8 h[2]; };

__device__ __forceinline__ float silu(float x) {
  return x * (1.0f / (1.0f + __expf(-x)));
}

// A fragment (16x32 f16, M = lane%16) from row-major f16 in LDS.
// halfs 0..7  -> K = kt*32 + hi*8 + j      (contiguous, one b128)
// halfs 8..15 -> K = kt*32 + 16 + hi*8 + j (contiguous, one b128)
__device__ __forceinline__ v16h load_a_frag(const _Float16* base, int stride, int kt) {
  int lane = threadIdx.x & 31;
  int m = lane & 15, hi = lane >> 4;
  const _Float16* p = base + m * stride + kt * 32 + hi * 8;
  V16 u;
  u.h[0] = *(const half8*)p;
  u.h[1] = *(const half8*)(p + 16);
  return u.v;
}

// B fragment (32x16 f16) pre-packed per fragment: [frag][lane][16 halfs]
// lane: N = lane%16, K = (lane/16)*8 + (j&7) + (j>>3)*16  -> 32 contiguous bytes per lane.
__device__ __forceinline__ v16h load_b_frag(const _Float16* wf, int frag) {
  int lane = threadIdx.x & 31;
  const _Float16* p = wf + ((size_t)frag << 9) + (lane << 4);
  V16 u;
  u.h[0] = *(const half8*)p;
  u.h[1] = *(const half8*)(p + 8);
  return u.v;
}

__device__ __forceinline__ v8f wmma16(v16h a, v16h b, v8f c) {
  return __builtin_amdgcn_wmma_f32_16x16x32_f16(
      /*neg_a=*/false, a, /*neg_b=*/false, b,
      /*c_mod=*/(short)0, c, /*reuse_a=*/false, /*reuse_b=*/false);
}

// ---------------- weight pre-pack: f32 [K x N] -> fragment-ordered f16 ----------------
__global__ void k_conv_w(const float* __restrict__ src, _Float16* __restrict__ dst,
                         int K, int Nn) {
  int ntiles = Nn >> 4;
  int total = (K >> 5) * ntiles * 512;
  for (int i = blockIdx.x * blockDim.x + threadIdx.x; i < total;
       i += gridDim.x * blockDim.x) {
    int f = i >> 9, r = i & 511;
    int lane = r >> 4, j = r & 15;
    int kt = f / ntiles, nt = f - kt * ntiles;
    int n = nt * 16 + (lane & 15);
    int k = kt * 32 + (lane >> 4) * 8 + (j & 7) + (j >> 3) * 16;
    dst[i] = (_Float16)src[(size_t)k * Nn + n];
  }
}

// ---------------- diffusion schedule + time embedding ----------------
__global__ void k_sched(const int* __restrict__ t, float* __restrict__ sqab,
                        float* __restrict__ sq1m, float* __restrict__ tem) {
  int tid = threadIdx.x;
  if (tid < NB) {
    int tt = t[tid];
    float prod = 1.0f;
    for (int i = 0; i < NUMT; ++i) {
      float beta = 1e-4f + (0.02f - 1e-4f) * (float)i / 99.0f;
      if (i <= tt) prod *= (1.0f - beta);
    }
    sqab[tid] = sqrtf(prod);
    sq1m[tid] = sqrtf(1.0f - prod);
  }
  for (int idx = tid; idx < NB * NTEMB; idx += blockDim.x) {
    int b = idx >> 5, k = idx & 31;
    int half = NTEMB / 2;
    int kk = (k < half) ? k : (k - half);
    float freq = __expf(-logf(10000.0f) * (float)kk / (float)(half - 1));
    float ang = (float)t[b] * freq;
    tem[idx] = (k < half) ? __sinf(ang) : __cosf(ang);
  }
}

__device__ __forceinline__ float blk_sum(float v, float* red, int a) {
  red[a] = v;
  __syncthreads();
  for (int s = 128; s > 0; s >>= 1) {
    if (a < s) red[a] += red[a + s];
    __syncthreads();
  }
  float r = red[0];
  __syncthreads();
  return r;
}

// center noise, build xt, center xt  (one block per batch, 256 threads = atoms)
__global__ void k_prep(const float* __restrict__ x0, const float* __restrict__ noise,
                       const float* __restrict__ sqab, const float* __restrict__ sq1m,
                       float* __restrict__ noise_c, float* __restrict__ xt) {
  __shared__ float red[256];
  int b = blockIdx.x, a = threadIdx.x;
  int n = b * NA + a;
  float nc[3], xv[3];
  for (int d = 0; d < 3; ++d) nc[d] = noise[n * 3 + d];
  for (int d = 0; d < 3; ++d) {
    float m = blk_sum(nc[d], red, a) * (1.0f / NA);
    nc[d] -= m;
  }
  float sa = sqab[b], s1 = sq1m[b];
  for (int d = 0; d < 3; ++d) xv[d] = sa * x0[n * 3 + d] + s1 * nc[d];
  for (int d = 0; d < 3; ++d) {
    float m = blk_sum(xv[d], red, a) * (1.0f / NA);
    xv[d] -= m;
  }
  for (int d = 0; d < 3; ++d) {
    noise_c[n * 3 + d] = nc[d];
    xt[n * 3 + d] = xv[d];
  }
}

// ---------------- node embedding GEMM: [N,96] @ [96,256] + b ----------------
__global__ void k_emb(const float* __restrict__ h, const float* __restrict__ tem,
                      const _Float16* __restrict__ wf, const float* __restrict__ bias,
                      float* __restrict__ hh) {
  __shared__ __align__(16) _Float16 bufA[WPB][16 * 96];
  int w = threadIdx.x >> 5, lane = threadIdx.x & 31;
  int tile = blockIdx.x * WPB + w;
  int n0 = tile * 16;
  {
    int m = lane >> 1, part = lane & 1;
    int node = n0 + m;
    int b = node >> 8;
    for (int kk = 0; kk < 48; ++kk) {
      int k = part * 48 + kk;
      float v = (k < NF) ? h[(size_t)node * NF + k] : tem[b * NTEMB + (k - NF)];
      bufA[w][m * 96 + k] = (_Float16)v;
    }
  }
  __syncthreads();
  int hi = lane >> 4, nl = lane & 15;
  for (int nt = 0; nt < 16; ++nt) {
    v8f acc = {};
    for (int kt = 0; kt < 3; ++kt) {
      v16h a = load_a_frag(&bufA[w][0], 96, kt);
      v16h bf = load_b_frag(wf, kt * 16 + nt);
      acc = wmma16(a, bf, acc);
    }
    int nn = nt * 16 + nl;
    float bb = bias[nn];
    for (int r = 0; r < 8; ++r) {
      int m = r + 8 * hi;
      hh[(size_t)(n0 + m) * NH + nn] = acc[r] + bb;
    }
  }
}

// ---------------- per-layer init: zero m_i, copy x ----------------
__global__ void k_init_layer(const float* __restrict__ x_src, float* __restrict__ x_dst,
                             float* __restrict__ m_i) {
  int i = blockIdx.x * blockDim.x + threadIdx.x;
  if (i < NN * NH) m_i[i] = 0.0f;
  if (i < NN * 3) x_dst[i] = x_src[i];
}

// ---------------- fused edge pipeline (3 chained WMMA GEMMs per 16-edge tile) ----------------
__global__ void __launch_bounds__(WPB * 32)
k_edge(const float* __restrict__ hh, const float* __restrict__ x,
       const int* __restrict__ row, const int* __restrict__ col,
       const _Float16* __restrict__ wfe1, const float* __restrict__ e1last,
       const float* __restrict__ eb1,
       const _Float16* __restrict__ wfe2, const float* __restrict__ eb2,
       const _Float16* __restrict__ wfc1, const float* __restrict__ cb1,
       const float* __restrict__ cw2,
       float* __restrict__ x_acc, float* __restrict__ m_i) {
  __shared__ __align__(16) _Float16 bufA[WPB][16 * 512];  // edge_in, later m_ij
  __shared__ __align__(16) _Float16 bufB[WPB][16 * 256];  // m
  __shared__ float cd[WPB][16][3];
  __shared__ float rad[WPB][16];
  __shared__ float wv[WPB][16];
  __shared__ int ridx[WPB][16], cidx[WPB][16];

  int w = threadIdx.x >> 5, lane = threadIdx.x & 31;
  int tile = blockIdx.x * WPB + w;
  int e0 = tile * 16;
  int hi = lane >> 4, nl = lane & 15;

  if (lane < 16) {
    int e = e0 + lane;
    int r = row[e], c = col[e];
    ridx[w][lane] = r;
    cidx[w][lane] = c;
    float s = 0.0f;
    for (int d = 0; d < 3; ++d) {
      float df = x[r * 3 + d] - x[c * 3 + d];
      cd[w][lane][d] = df;
      s += df * df;
    }
    rad[w][lane] = s;
  }
  __syncthreads();

  // edge_in tile: [16 edges] x [h[row] (256) | h[col] (256)] as f16
  {
    int m = lane >> 1, part = lane & 1;
    int node = part ? cidx[w][m] : ridx[w][m];
    const float4* src = (const float4*)(hh + (size_t)node * NH);
    _Float16* dst = &bufA[w][m * 512 + part * 256];
    for (int k4 = 0; k4 < 64; ++k4) {
      float4 v = src[k4];
      dst[k4 * 4 + 0] = (_Float16)v.x;
      dst[k4 * 4 + 1] = (_Float16)v.y;
      dst[k4 * 4 + 2] = (_Float16)v.z;
      dst[k4 * 4 + 3] = (_Float16)v.w;
    }
  }
  __syncthreads();

  // GEMM1: edge_in @ eW1[0:512]  (+ radial * eW1[512] + eb1), SiLU -> bufB (m)
  for (int nt = 0; nt < 16; ++nt) {
    v8f acc = {};
    for (int kt = 0; kt < 16; ++kt) {
      v16h a = load_a_frag(&bufA[w][0], 512, kt);
      v16h bf = load_b_frag(wfe1, kt * 16 + nt);
      acc = wmma16(a, bf, acc);
    }
    int n = nt * 16 + nl;
    float wl = e1last[n], bb = eb1[n];
    for (int r = 0; r < 8; ++r) {
      int m = r + 8 * hi;
      bufB[w][m * 256 + n] = (_Float16)silu(acc[r] + rad[w][m] * wl + bb);
    }
  }
  __syncthreads();

  // GEMM2: m @ eW2 + eb2, SiLU -> m_ij (stored in bufA, row stride 512, cols 0..255)
  for (int nt = 0; nt < 16; ++nt) {
    v8f acc = {};
    for (int kt = 0; kt < 8; ++kt) {
      v16h a = load_a_frag(&bufB[w][0], 256, kt);
      v16h bf = load_b_frag(wfe2, kt * 16 + nt);
      acc = wmma16(a, bf, acc);
    }
    int n = nt * 16 + nl;
    float bb = eb2[n];
    for (int r = 0; r < 8; ++r) {
      int m = r + 8 * hi;
      bufA[w][m * 512 + n] = (_Float16)silu(acc[r] + bb);
    }
  }
  __syncthreads();

  // GEMM3: m_ij @ cW1 + cb1, SiLU, folded dot with cW2 -> per-edge w
  float wpart[8];
  for (int r = 0; r < 8; ++r) wpart[r] = 0.0f;
  for (int nt = 0; nt < 16; ++nt) {
    v8f acc = {};
    for (int kt = 0; kt < 8; ++kt) {
      v16h a = load_a_frag(&bufA[w][0], 512, kt);
      v16h bf = load_b_frag(wfc1, kt * 16 + nt);
      acc = wmma16(a, bf, acc);
    }
    int n = nt * 16 + nl;
    float bb = cb1[n], c2 = cw2[n];
    for (int r = 0; r < 8; ++r) wpart[r] += silu(acc[r] + bb) * c2;
  }
  for (int off = 1; off < 16; off <<= 1)
    for (int r = 0; r < 8; ++r) wpart[r] += __shfl_xor(wpart[r], off, 32);
  if (nl == 0)
    for (int r = 0; r < 8; ++r) wv[w][r + 8 * hi] = wpart[r];
  __syncthreads();

  // scatter: coord updates (atomic into x_acc[row]) + m_i segment-sum
  if (lane < 16) {
    int r = ridx[w][lane];
    float ww = wv[w][lane];
    for (int d = 0; d < 3; ++d)
      atomicAdd(&x_acc[r * 3 + d], cd[w][lane][d] * ww);
  }
  for (int idx = lane; idx < 16 * 256; idx += 32) {
    int m = idx >> 8, n = idx & 255;
    atomicAdd(&m_i[(size_t)ridx[w][m] * NH + n], (float)bufA[w][m * 512 + n]);
  }
}

// ---------------- node MLP: h_new = h + MLP(concat(h, m_i)) ----------------
__global__ void __launch_bounds__(WPB * 32)
k_node(const float* __restrict__ hh, const float* __restrict__ m_i,
       const _Float16* __restrict__ wfn1, const float* __restrict__ nb1,
       const _Float16* __restrict__ wfn2, const float* __restrict__ nb2,
       float* __restrict__ hh_out) {
  __shared__ __align__(16) _Float16 bufA[WPB][16 * 512];
  __shared__ __align__(16) _Float16 bufB[WPB][16 * 256];
  int w = threadIdx.x >> 5, lane = threadIdx.x & 31;
  int tile = blockIdx.x * WPB + w;
  int n0 = tile * 16;
  int hi = lane >> 4, nl = lane & 15;
  {
    int m = lane >> 1, part = lane & 1;
    const float* srcp = (part ? m_i : hh) + (size_t)(n0 + m) * NH;
    const float4* src = (const float4*)srcp;
    _Float16* dst = &bufA[w][m * 512 + part * 256];
    for (int k4 = 0; k4 < 64; ++k4) {
      float4 v = src[k4];
      dst[k4 * 4 + 0] = (_Float16)v.x;
      dst[k4 * 4 + 1] = (_Float16)v.y;
      dst[k4 * 4 + 2] = (_Float16)v.z;
      dst[k4 * 4 + 3] = (_Float16)v.w;
    }
  }
  __syncthreads();
  for (int nt = 0; nt < 16; ++nt) {
    v8f acc = {};
    for (int kt = 0; kt < 16; ++kt) {
      v16h a = load_a_frag(&bufA[w][0], 512, kt);
      v16h bf = load_b_frag(wfn1, kt * 16 + nt);
      acc = wmma16(a, bf, acc);
    }
    int n = nt * 16 + nl;
    float bb = nb1[n];
    for (int r = 0; r < 8; ++r) {
      int m = r + 8 * hi;
      bufB[w][m * 256 + n] = (_Float16)silu(acc[r] + bb);
    }
  }
  __syncthreads();
  for (int nt = 0; nt < 16; ++nt) {
    v8f acc = {};
    for (int kt = 0; kt < 8; ++kt) {
      v16h a = load_a_frag(&bufB[w][0], 256, kt);
      v16h bf = load_b_frag(wfn2, kt * 16 + nt);
      acc = wmma16(a, bf, acc);
    }
    int n = nt * 16 + nl;
    float bb = nb2[n];
    for (int r = 0; r < 8; ++r) {
      int m = r + 8 * hi;
      size_t o = (size_t)(n0 + m) * NH + n;
      hh_out[o] = hh[o] + acc[r] + bb;
    }
  }
}

// ---------------- loss ----------------
__global__ void k_zero1(float* p) {
  if (threadIdx.x == 0 && blockIdx.x == 0) p[0] = 0.0f;
}

__global__ void k_loss(const float* __restrict__ xfin, const float* __restrict__ xt,
                       const float* __restrict__ noise_c, float* __restrict__ out) {
  __shared__ float red[256];
  int b = blockIdx.x, a = threadIdx.x;
  int n = b * NA + a;
  float dp[3];
  for (int d = 0; d < 3; ++d) dp[d] = xfin[n * 3 + d] - xt[n * 3 + d];
  for (int d = 0; d < 3; ++d) {
    float m = blk_sum(dp[d], red, a) * (1.0f / NA);
    dp[d] -= m;
  }
  float s = 0.0f;
  for (int d = 0; d < 3; ++d) {
    float df = dp[d] - noise_c[n * 3 + d];
    s += df * df;
  }
  float bs = blk_sum(s, red, a);
  if (a == 0) atomicAdd(out, bs / (float)(NN * 3));
}

// ---------------- host launcher ----------------
extern "C" void kernel_launch(void* const* d_in, const int* in_sizes, int n_in,
                              void* d_out, int out_size, void* d_ws, size_t ws_size,
                              hipStream_t stream) {
  (void)in_sizes; (void)n_in; (void)out_size; (void)ws_size;
  const float* h     = (const float*)d_in[0];
  const float* x0    = (const float*)d_in[1];
  const float* noise = (const float*)d_in[2];
  const int*   t     = (const int*)d_in[3];
  const int*   row   = (const int*)d_in[4];
  const int*   col   = (const int*)d_in[5];
  const float* W_emb = (const float*)d_in[6];
  const float* b_emb = (const float*)d_in[7];
  const float* eW1   = (const float*)d_in[8];
  const float* eb1   = (const float*)d_in[9];
  const float* eW2   = (const float*)d_in[10];
  const float* eb2   = (const float*)d_in[11];
  const float* cW1   = (const float*)d_in[12];
  const float* cb1   = (const float*)d_in[13];
  const float* cW2   = (const float*)d_in[14];
  const float* nW1   = (const float*)d_in[15];
  const float* nb1   = (const float*)d_in[16];
  const float* nW2   = (const float*)d_in[17];
  const float* nb2   = (const float*)d_in[18];
  float* out = (float*)d_out;

  char* ws = (char*)d_ws;
  size_t off = 0;
  auto ALLOC = [&](size_t bytes) -> char* {
    char* p = ws + off;
    off += (bytes + 255) & ~(size_t)255;
    return p;
  };

  const size_t FR_E1 = 16 * 16 * 512;  // frags for 512x256
  const size_t FR_SQ = 8 * 16 * 512;   // frags for 256x256
  _Float16* wf_emb = (_Float16*)ALLOC(3 * 16 * 512 * 2);
  _Float16* wf_e1  = (_Float16*)ALLOC(NL * FR_E1 * 2);
  _Float16* wf_e2  = (_Float16*)ALLOC(NL * FR_SQ * 2);
  _Float16* wf_c1  = (_Float16*)ALLOC(NL * FR_SQ * 2);
  _Float16* wf_n1  = (_Float16*)ALLOC(NL * FR_E1 * 2);
  _Float16* wf_n2  = (_Float16*)ALLOC(NL * FR_SQ * 2);
  float* tem     = (float*)ALLOC(NB * NTEMB * 4);
  float* sqab    = (float*)ALLOC(NB * 4);
  float* sq1m    = (float*)ALLOC(NB * 4);
  float* noise_c = (float*)ALLOC(NN * 3 * 4);
  float* xt      = (float*)ALLOC(NN * 3 * 4);
  float* xb0     = (float*)ALLOC(NN * 3 * 4);
  float* xb1     = (float*)ALLOC(NN * 3 * 4);
  float* hb0     = (float*)ALLOC((size_t)NN * NH * 4);
  float* hb1     = (float*)ALLOC((size_t)NN * NH * 4);
  float* m_i     = (float*)ALLOC((size_t)NN * NH * 4);

  auto conv = [&](const float* src, _Float16* dst, int K, int Nn) {
    int total = (K / 32) * (Nn / 16) * 512;
    k_conv_w<<<(total + 255) / 256, 256, 0, stream>>>(src, dst, K, Nn);
  };
  conv(W_emb, wf_emb, 96, 256);
  for (int l = 0; l < NL; ++l) {
    conv(eW1 + (size_t)l * 513 * 256, wf_e1 + l * FR_E1, 512, 256);  // rows 0..511
    conv(eW2 + (size_t)l * 256 * 256, wf_e2 + l * FR_SQ, 256, 256);
    conv(cW1 + (size_t)l * 256 * 256, wf_c1 + l * FR_SQ, 256, 256);
    conv(nW1 + (size_t)l * 512 * 256, wf_n1 + l * FR_E1, 512, 256);
    conv(nW2 + (size_t)l * 256 * 256, wf_n2 + l * FR_SQ, 256, 256);
  }

  k_sched<<<1, 256, 0, stream>>>(t, sqab, sq1m, tem);
  k_prep<<<NB, 256, 0, stream>>>(x0, noise, sqab, sq1m, noise_c, xt);

  // embedding: 256 M-tiles, 2 waves/block
  k_emb<<<(NN / 16) / WPB, WPB * 32, 0, stream>>>(h, tem, wf_emb, b_emb, hb0);

  for (int l = 0; l < NL; ++l) {
    const float* hin  = (l & 1) ? hb1 : hb0;
    float*       hout = (l & 1) ? hb0 : hb1;
    const float* xsrc = (l == 0) ? xt : ((l & 1) ? xb0 : xb1);
    float*       xdst = (l & 1) ? xb1 : xb0;

    k_init_layer<<<(NN * NH + 255) / 256, 256, 0, stream>>>(xsrc, xdst, m_i);

    k_edge<<<(NE / 16) / WPB, WPB * 32, 0, stream>>>(
        hin, xsrc, row, col,
        wf_e1 + l * FR_E1, eW1 + (size_t)l * 513 * 256 + (size_t)512 * 256,
        eb1 + l * NH,
        wf_e2 + l * FR_SQ, eb2 + l * NH,
        wf_c1 + l * FR_SQ, cb1 + l * NH, cW2 + l * NH,
        xdst, m_i);

    k_node<<<(NN / 16) / WPB, WPB * 32, 0, stream>>>(
        hin, m_i, wf_n1 + l * FR_E1, nb1 + l * NH,
        wf_n2 + l * FR_SQ, nb2 + l * NH, hout);
  }

  k_zero1<<<1, 32, 0, stream>>>(out);
  k_loss<<<NB, 256, 0, stream>>>(xb1, xt, noise_c, out);
}
